// Attention_28999619183038
// MI455X (gfx1250) — compile-verified, matrix operand order checked by reading, not run on hardware
//
#include <hip/hip_runtime.h>
#include <math.h>

// ---------------------------------------------------------------- types ----
typedef _Float16 half_t;
typedef __attribute__((ext_vector_type(16))) _Float16     v16h;
typedef __attribute__((ext_vector_type(8)))  _Float16     v8h;
typedef __attribute__((ext_vector_type(8)))  float        v8f;
typedef __attribute__((ext_vector_type(4)))  float        v4f;
typedef __attribute__((ext_vector_type(4)))  unsigned int u32x4;
typedef __attribute__((ext_vector_type(8)))  int          i32x8;
typedef __attribute__((ext_vector_type(4)))  int          i32x4;

constexpr int kB      = 2;
constexpr int kN      = 2048;
constexpr int kDim    = 2048;
constexpr int kH      = 32;
constexpr int kD      = 64;
constexpr int kTriple = 3 * kDim;   // 6144

#if __has_builtin(__builtin_amdgcn_tensor_load_to_lds)
#define USE_TDM 1
#endif

// D = A*B + C, 16x16x32 f16 -> f32 accumulate
__device__ __forceinline__ v8f wmma16(v16h a, v16h b, v8f c) {
  return __builtin_amdgcn_wmma_f32_16x16x32_f16(
      /*neg_a=*/false, a, /*neg_b=*/false, b,
      /*c_mod=*/(short)0, c, /*reuse_a=*/false, /*reuse_b=*/false);
}

// Assemble a 16-bit A/B WMMA fragment for a K-major row (row*ld + k layout).
// Per ISA layout: lanes 0-15 hold K = kb+{0..7, 16..23}, lanes 16-31 hold
// K = kb+{8..15, 24..31}.  Two 16-byte loads (global_load_b128 / ds_load_b128).
__device__ __forceinline__ v16h load_frag(const half_t* p0, int ld, int row,
                                          int kb, int grp) {
  const half_t* p = p0 + (long)row * ld + kb + 8 * grp;
  union { v16h v; struct { v8h lo; v8h hi; } s; } u;
  u.s.lo = *(const v8h*)p;
  u.s.hi = *(const v8h*)(p + 16);
  return u.v;
}

// ------------------------------------------------------- fp32 -> fp16 ------
__global__ void __launch_bounds__(256) cvt_f32_f16(const float* __restrict__ in,
                                                   half_t* __restrict__ out,
                                                   long n) {
  long i = ((long)blockIdx.x * blockDim.x + threadIdx.x) * 8;
  if (i + 8 > n) return;
  v4f a = *(const v4f*)(in + i);
  v4f b = *(const v4f*)(in + i + 4);
  v8h o;
  o[0] = (half_t)a[0]; o[1] = (half_t)a[1]; o[2] = (half_t)a[2]; o[3] = (half_t)a[3];
  o[4] = (half_t)b[0]; o[5] = (half_t)b[1]; o[6] = (half_t)b[2]; o[7] = (half_t)b[3];
  *(v8h*)(out + i) = o;
}

// ------------------------------------------- C = A * B^T  (f16 in/out) -----
// A: [M,K] row-major f16.  Bm: [Nc,K] row-major f16 (i.e. B^T logical).
// Block: 256 thr = 8 waves, tile 256 rows x 64 cols, each wave 32x64
// (two A fragments reuse each B fragment -> ~341 FLOP/byte from cache).
__global__ void __launch_bounds__(256) gemm_f16(const half_t* __restrict__ A,
                                                const half_t* __restrict__ Bm,
                                                half_t* __restrict__ C,
                                                int M, int Nc, int K) {
  const int lane = threadIdx.x & 31;
  const int wave = threadIdx.x >> 5;
  const int grp  = lane >> 4;
  const int lc   = lane & 15;
  const int row0 = blockIdx.y * 256 + wave * 32;
  const int col0 = blockIdx.x * 64;

  v8f acc[8] = {};
  for (int kb = 0; kb < K; kb += 32) {
    v16h a0 = load_frag(A, K, row0 + lc,      kb, grp);
    v16h a1 = load_frag(A, K, row0 + 16 + lc, kb, grp);
#pragma unroll
    for (int t = 0; t < 4; ++t) {
      v16h b = load_frag(Bm, K, col0 + t * 16 + lc, kb, grp);
      acc[t]     = wmma16(a0, b, acc[t]);
      acc[4 + t] = wmma16(a1, b, acc[4 + t]);
    }
  }
#pragma unroll
  for (int hr = 0; hr < 2; ++hr)
#pragma unroll
    for (int t = 0; t < 4; ++t)
#pragma unroll
      for (int r = 0; r < 8; ++r) {
        int row = row0 + hr * 16 + r + 8 * grp;
        int col = col0 + t * 16 + lc;
        C[(long)row * Nc + col] = (half_t)acc[hr * 4 + t][r];
      }
}

// ------------------------------- out GEMM with row mask, f32 epilogue ------
__global__ void __launch_bounds__(256) gemm_out(const half_t* __restrict__ A,
                                                const half_t* __restrict__ Bm,
                                                const unsigned char* __restrict__ mask,
                                                float* __restrict__ C,
                                                int M, int Nc, int K) {
  const int lane = threadIdx.x & 31;
  const int wave = threadIdx.x >> 5;
  const int grp  = lane >> 4;
  const int lc   = lane & 15;
  const int row0 = blockIdx.y * 256 + wave * 32;
  const int col0 = blockIdx.x * 64;

  v8f acc[8] = {};
  for (int kb = 0; kb < K; kb += 32) {
    v16h a0 = load_frag(A, K, row0 + lc,      kb, grp);
    v16h a1 = load_frag(A, K, row0 + 16 + lc, kb, grp);
#pragma unroll
    for (int t = 0; t < 4; ++t) {
      v16h b = load_frag(Bm, K, col0 + t * 16 + lc, kb, grp);
      acc[t]     = wmma16(a0, b, acc[t]);
      acc[4 + t] = wmma16(a1, b, acc[4 + t]);
    }
  }
#pragma unroll
  for (int hr = 0; hr < 2; ++hr)
#pragma unroll
    for (int t = 0; t < 4; ++t)
#pragma unroll
      for (int r = 0; r < 8; ++r) {
        int row = row0 + hr * 16 + r + 8 * grp;
        int col = col0 + t * 16 + lc;
        float mv = mask[row] ? 1.0f : 0.0f;
        C[(long)row * Nc + col] = acc[hr * 4 + t][r] * mv;
      }
}

// -------------------------- per-head LayerNorm + RoPE + head split ---------
__device__ __forceinline__ void ln_rope_one(const half_t* __restrict__ src,
                                            const float* __restrict__ w,
                                            const float* __restrict__ bia,
                                            float cs, float sn, int lane,
                                            half_t* __restrict__ dst) {
  float t0 = (float)src[lane];
  float t1 = (float)src[lane + 32];
  float s = t0 + t1;
#pragma unroll
  for (int off = 1; off < 32; off <<= 1) s += __shfl_xor(s, off, 32);
  float m  = s * (1.0f / 64.0f);
  float e0 = t0 - m, e1 = t1 - m;
  float v = e0 * e0 + e1 * e1;
#pragma unroll
  for (int off = 1; off < 32; off <<= 1) v += __shfl_xor(v, off, 32);
  float rstd = rsqrtf(v * (1.0f / 64.0f) + 1e-6f);
  float n0 = e0 * rstd * w[lane]      + bia[lane];
  float n1 = e1 * rstd * w[lane + 32] + bia[lane + 32];
  // RoPE on dims [0,32): rotate-half partner is lane^16
  float part = __shfl_xor(n0, 16, 32);
  float r0 = (lane < 16) ? (n0 * cs - part * sn) : (n0 * cs + part * sn);
  dst[lane]      = (half_t)r0;
  dst[lane + 32] = (half_t)n1;   // pass-through dims [32,64)
}

// one wave per (b,n,h); 8 waves / block
__global__ void __launch_bounds__(256) ln_rope(const half_t* __restrict__ qkv,
                                               const float* __restrict__ qw,
                                               const float* __restrict__ qb,
                                               const float* __restrict__ kw,
                                               const float* __restrict__ kbi,
                                               const float* __restrict__ freqs,
                                               half_t* __restrict__ qh,
                                               half_t* __restrict__ kh,
                                               half_t* __restrict__ vh) {
  const int lane = threadIdx.x & 31;
  const int wave = threadIdx.x >> 5;
  const int gid  = blockIdx.x * 8 + wave;      // 0 .. B*N*H-1
  const int h = gid & (kH - 1);
  const int n = (gid >> 5) & (kN - 1);
  const int b = gid >> 16;                     // / (kN*kH)

  const long src = (long)(b * kN + n) * kTriple + h * kD;
  const long dst = ((long)(b * kH + h) * kN + n) * kD;

  float f  = freqs[n * 32 + lane];
  float cs = cosf(f), sn = sinf(f);

  ln_rope_one(qkv + src,        qw, qb,  cs, sn, lane, qh + dst);
  ln_rope_one(qkv + src + kDim, kw, kbi, cs, sn, lane, kh + dst);
  vh[dst + lane]      = qkv[src + 2 * kDim + lane];
  vh[dst + lane + 32] = qkv[src + 2 * kDim + lane + 32];
}

// --------------------------------------- causal masked flash attention -----
// grid (N/128, H, B); block 256 = 8 waves; wave owns 16 query rows.
// K tile staged into LDS by the Tensor Data Mover (async DMA, TENSORcnt);
// V tile staged manually transposed so PV B-fragments are contiguous B128.
__global__ void __launch_bounds__(256) flash_attn(const half_t* __restrict__ Qg,
                                                  const half_t* __restrict__ Kg,
                                                  const half_t* __restrict__ Vg,
                                                  const unsigned char* __restrict__ mask,
                                                  half_t* __restrict__ Og) {
  __shared__ __attribute__((aligned(16))) half_t Ks[64 * 64];     // row-major [key][d]
  __shared__ __attribute__((aligned(16))) half_t Vt[64 * 64];     // transposed [d][key]
  __shared__ __attribute__((aligned(16))) half_t Ps[8][16 * 64];  // per-wave P scratch

  const int lane = threadIdx.x & 31;
  const int wave = threadIdx.x >> 5;
  const int grp  = lane >> 4;
  const int lc   = lane & 15;
  const int qb = blockIdx.x, h = blockIdx.y, b = blockIdx.z;
  const long headoff = (long)(b * kH + h) * kN * kD;
  const int row0 = qb * 128 + wave * 16;

  // Q fragments, pre-scaled by 1/sqrt(D) = 1/8
  v16h qa0 = load_frag(Qg + headoff, kD, row0 + lc, 0, grp);
  v16h qa1 = load_frag(Qg + headoff, kD, row0 + lc, 32, grp);
#pragma unroll
  for (int i = 0; i < 16; ++i) {
    qa0[i] = (half_t)((float)qa0[i] * 0.125f);
    qa1[i] = (half_t)((float)qa1[i] * 0.125f);
  }

  v8f o[4] = {};
  float mrow[8], lrow[8];
#pragma unroll
  for (int r = 0; r < 8; ++r) { mrow[r] = -3.0e38f; lrow[r] = 0.0f; }

  const int nkb = 2 * qb + 2;   // causal: only key blocks <= query block
  for (int kb = 0; kb < nkb; ++kb) {
#ifdef USE_TDM
    // --- K tile via Tensor Data Mover: 2D D#, 64x64 f16, row-major ---
    if (wave == 0) {
      unsigned long long ga =
          (unsigned long long)(const void*)(Kg + headoff + (long)kb * 64 * kD);
      unsigned ldsa = (unsigned)(unsigned long long)(void*)&Ks[0];
      // group0: count=1 | lds_addr | global_addr | type=2
      u32x4 g0 = { 1u, ldsa, (unsigned)ga,
                   (unsigned)((ga >> 32) & 0x01FFFFFFu) | 0x80000000u };
      // group1: data_size=2B; tensor_dim0=64; tensor_dim1=2048;
      //         tile_dim0=64; tile_dim1=64; dim0_stride=64
      i32x8 g1 = { 0x10000, 64 << 16, 2048 << 16, 64 << 16, 64, 64, 0, 0 };
      i32x4 gz = { 0, 0, 0, 0 };
#if defined(__clang_major__) && __clang_major__ >= 23
      i32x8 gz8 = { 0, 0, 0, 0, 0, 0, 0, 0 };
      __builtin_amdgcn_tensor_load_to_lds(g0, g1, gz, gz, gz8, 0);
#else
      __builtin_amdgcn_tensor_load_to_lds(g0, g1, gz, gz, 0);
#endif
      __builtin_amdgcn_s_wait_tensorcnt(0);
    }
#else
    // --- fallback: manual K staging ---
    for (int i = threadIdx.x; i < 64 * 8; i += 256) {
      int r = i >> 3, c8 = (i & 7) * 8;
      *(v8h*)&Ks[r * 64 + c8] =
          *(const v8h*)(Kg + headoff + (long)(kb * 64 + r) * kD + c8);
    }
#endif
    // --- V tile staged transposed (TDM cannot transpose) ---
    for (int i = threadIdx.x; i < 64 * 8; i += 256) {
      int r = i >> 3, c8 = (i & 7) * 8;
      v8h vv = *(const v8h*)(Vg + headoff + (long)(kb * 64 + r) * kD + c8);
#pragma unroll
      for (int j = 0; j < 8; ++j) Vt[(c8 + j) * 64 + r] = vv[j];
    }
    if (kb + 1 < nkb)
      __builtin_prefetch(Vg + headoff + (long)((kb + 1) * 64) * kD, 0, 1);
    __syncthreads();

    // S = (Q/8) K^T : 4 col tiles x 2 k-frags = 8 WMMAs
    v8f s[4] = {};
#pragma unroll
    for (int t = 0; t < 4; ++t) {
      v16h bk0 = load_frag(Ks, 64, t * 16 + lc, 0, grp);
      v16h bk1 = load_frag(Ks, 64, t * 16 + lc, 32, grp);
      s[t] = wmma16(qa0, bk0, s[t]);
      s[t] = wmma16(qa1, bk1, s[t]);
    }

    // key-padding + causal mask
#pragma unroll
    for (int t = 0; t < 4; ++t) {
      int j = kb * 64 + t * 16 + lc;
      bool kvalid = mask[b * kN + j] != 0;
#pragma unroll
      for (int r = 0; r < 8; ++r) {
        int qi = row0 + r + 8 * grp;
        bool ok = kvalid && (j <= qi);
        s[t][r] = ok ? s[t][r] : -3.0e38f;
      }
    }

    // online softmax (rows live in 16-lane groups)
    float alpha[8];
#pragma unroll
    for (int r = 0; r < 8; ++r) {
      float rm = fmaxf(fmaxf(s[0][r], s[1][r]), fmaxf(s[2][r], s[3][r]));
#pragma unroll
      for (int off = 1; off < 16; off <<= 1) rm = fmaxf(rm, __shfl_xor(rm, off, 32));
      float mn = fmaxf(mrow[r], rm);
      alpha[r] = __expf(mrow[r] - mn);
      mrow[r] = mn;
      float rs = 0.0f;
#pragma unroll
      for (int t = 0; t < 4; ++t) {
        float p = __expf(s[t][r] - mn);
        s[t][r] = p;
        rs += p;
      }
#pragma unroll
      for (int off = 1; off < 16; off <<= 1) rs += __shfl_xor(rs, off, 32);
      lrow[r] = lrow[r] * alpha[r] + rs;
    }
#pragma unroll
    for (int ct = 0; ct < 4; ++ct)
#pragma unroll
      for (int r = 0; r < 8; ++r) o[ct][r] *= alpha[r];

    // P: C-layout f32 -> A-layout f16 via per-wave LDS round trip
    half_t* pw = &Ps[wave][0];
#pragma unroll
    for (int t = 0; t < 4; ++t)
#pragma unroll
      for (int r = 0; r < 8; ++r)
        pw[(r + 8 * grp) * 64 + t * 16 + lc] = (half_t)s[t][r];
    asm volatile("s_wait_dscnt 0" ::: "memory");

    v16h pa0 = load_frag(pw, 64, lc, 0, grp);
    v16h pa1 = load_frag(pw, 64, lc, 32, grp);

    // O += P V : 4 col tiles x 2 k-frags = 8 WMMAs
#pragma unroll
    for (int ct = 0; ct < 4; ++ct) {
      v16h bv0 = load_frag(Vt, 64, ct * 16 + lc, 0, grp);
      v16h bv1 = load_frag(Vt, 64, ct * 16 + lc, 32, grp);
      o[ct] = wmma16(pa0, bv0, o[ct]);
      o[ct] = wmma16(pa1, bv1, o[ct]);
    }
    __syncthreads();
  }

  // normalize + write merged-head layout [B, N, H*D] (f16 for the out GEMM)
  float invl[8];
#pragma unroll
  for (int r = 0; r < 8; ++r) invl[r] = (lrow[r] > 0.0f) ? (1.0f / lrow[r]) : 0.0f;
#pragma unroll
  for (int ct = 0; ct < 4; ++ct)
#pragma unroll
    for (int r = 0; r < 8; ++r) {
      int qi = row0 + r + 8 * grp;
      int d  = ct * 16 + lc;
      Og[(long)(b * kN + qi) * kDim + h * kD + d] = (half_t)(o[ct][r] * invl[r]);
    }
}

// --------------------------------------------------------------- launch ----
extern "C" void kernel_launch(void* const* d_in, const int* in_sizes, int n_in,
                              void* d_out, int out_size, void* d_ws, size_t ws_size,
                              hipStream_t stream) {
  const float* x     = (const float*)d_in[0];
  const float* Wqkv  = (const float*)d_in[1];
  const float* Wout  = (const float*)d_in[2];
  const float* qlw   = (const float*)d_in[3];
  const float* qlb   = (const float*)d_in[4];
  const float* klw   = (const float*)d_in[5];
  const float* klb   = (const float*)d_in[6];
  const float* freqs = (const float*)d_in[7];
  const unsigned char* mask = (const unsigned char*)d_in[8];

  const long nTok = (long)kB * kN;                       // 4096
  char* ws = (char*)d_ws;
  half_t* xh   = (half_t*)ws;  ws += nTok * kDim * 2;            // 16 MB
  half_t* wqh  = (half_t*)ws;  ws += (long)kTriple * kDim * 2;   // 24 MB
  half_t* woh  = (half_t*)ws;  ws += (long)kDim * kDim * 2;      //  8 MB
  half_t* qkvh = (half_t*)ws;  ws += nTok * kTriple * 2;         // 48 MB
  half_t* qhp  = (half_t*)ws;  ws += (long)kB * kH * kN * kD * 2;
  half_t* khp  = (half_t*)ws;  ws += (long)kB * kH * kN * kD * 2;
  half_t* vhp  = (half_t*)ws;  ws += (long)kB * kH * kN * kD * 2;
  half_t* att  = (half_t*)ws;  ws += nTok * kDim * 2;            // 16 MB

  // 1) fp32 -> fp16 operand conversion
  {
    long n1 = nTok * kDim, n2 = (long)kTriple * kDim, n3 = (long)kDim * kDim;
    cvt_f32_f16<<<(int)(n1 / 2048), 256, 0, stream>>>(x,    xh,  n1);
    cvt_f32_f16<<<(int)(n2 / 2048), 256, 0, stream>>>(Wqkv, wqh, n2);
    cvt_f32_f16<<<(int)(n3 / 2048), 256, 0, stream>>>(Wout, woh, n3);
  }
  // 2) QKV projection: [4096,2048] x [6144,2048]^T
  gemm_f16<<<dim3(kTriple / 64, (int)(nTok / 256)), 256, 0, stream>>>(
      xh, wqh, qkvh, (int)nTok, kTriple, kDim);
  // 3) per-head LN + RoPE + head split
  ln_rope<<<(int)(kB * kN * kH / 8), 256, 0, stream>>>(
      qkvh, qlw, qlb, klw, klb, freqs, qhp, khp, vhp);
  // 4) causal masked flash attention (TDM-staged K tiles)
  flash_attn<<<dim3(kN / 128, kH, kB), 256, 0, stream>>>(
      qhp, khp, vhp, mask, att);
  // 5) output projection + padding mask, fp32 out
  gemm_out<<<dim3(kDim / 64, (int)(nTok / 256)), 256, 0, stream>>>(
      att, woh, mask, (float*)d_out, (int)nTok, kDim, kDim);
}
